// ET_17721035064071
// MI455X (gfx1250) — matrix operand order repaired and checked by reading, not compiled, not run on hardware
//
#include <hip/hip_runtime.h>

// ---------------------------------------------------------------------------
// Energy Transformer forward for MI455X (gfx1250, wave32, WMMA).
// All heavy math runs through v_wmma_f32_16x16x32_f16 (f16 in, f32 accum).
// Staging: async global->LDS (ASYNCcnt) when available, else 16B vector
// loads + ds_store_b128; double buffered; global_prefetch warms k+2.
// ---------------------------------------------------------------------------

typedef __attribute__((ext_vector_type(16))) _Float16 v16h;
typedef __attribute__((ext_vector_type(8)))  float    v8f;
typedef __attribute__((ext_vector_type(4)))  int      v4i;

#if __has_builtin(__builtin_amdgcn_global_load_async_to_lds_b128) && \
    __has_builtin(__builtin_amdgcn_s_wait_asynccnt)
#define HAS_ASYNC_LDS 1
#else
#define HAS_ASYNC_LDS 0
#endif

// builtin param types: (int4 __device__* /*AS1*/, int4 __shared__* /*AS3*/, Imm, Imm)
typedef __attribute__((address_space(1))) v4i glb_v4i;
typedef __attribute__((address_space(3))) v4i lds_v4i;

// 16-byte global -> LDS copy: async-to-LDS when the toolchain has it.
__device__ __forceinline__ void copy16_g2l(const _Float16* gsrc, _Float16* ldst) {
#if HAS_ASYNC_LDS
  __builtin_amdgcn_global_load_async_to_lds_b128(
      (glb_v4i*)gsrc, (lds_v4i*)ldst, 0, 0);
#else
  *(uint4*)ldst = *(const uint4*)gsrc;
#endif
}

__device__ __forceinline__ void async_wait_all() {
#if HAS_ASYNC_LDS
  __builtin_amdgcn_s_wait_asynccnt(0);
#endif
}

// ------------------------- model constants ---------------------------------
constexpr int CB   = 64;    // batch
constexpr int CN   = 196;   // patches
constexpr int CD   = 768;   // patch dim
constexpr int CTK  = 768;   // token dim
constexpr int CQK  = 64;    // per-head qk dim
constexpr int CH   = 12;    // heads
constexpr int CHN  = 3072;  // hopfield hidden
constexpr int CNP  = 197;   // N+1 (with CLS)
constexpr float CBETA = 0.125f;  // 1/sqrt(64)
constexpr long MFULL = (long)CB * CNP;   // 12608
constexpr long MENC  = (long)CB * CN;    // 12544

// ------------------------- GEMM tile config --------------------------------
#define BM 128
#define BN 128
#define BKT 32
#define LDAS 48   // padded LDS stride (halves); 96B, multiple of 16B
#define LDBS 48

union FragH { v16h v; uint4 q[2]; };

// C[M,N] = op(A,B) (+bias)(ReLU)(+=C)
// OPB==1: B given as W[N,K] row-major (C = A * W^T)
// OPB==0: B given as B[K,N] row-major (C = A * B)
// Batched via blockIdx.z: base += (z/idiv)*sXo + (z%idiv)*sXi
template <int OPB, typename TC, bool RELU, bool ACCUM>
__global__ __launch_bounds__(256) void gemm_wmma(
    const _Float16* __restrict__ A, const _Float16* __restrict__ Bm,
    TC* __restrict__ C, const float* __restrict__ bias,
    int M, int N, int K, int lda, int ldb, int ldc,
    long sAo, long sAi, long sBo, long sBi, long sCo, long sCi, int idiv)
{
  __shared__ __align__(16) _Float16 sA[2][BM * LDAS];
  __shared__ __align__(16) _Float16 sB[2][BN * LDBS];

  const int z = blockIdx.z;
  const int zo = z / idiv, zi = z % idiv;
  const _Float16* Ab = A + (long)zo * sAo + (long)zi * sAi;
  const _Float16* Bb = Bm + (long)zo * sBo + (long)zi * sBi;
  TC* Cb = C + (long)zo * sCo + (long)zi * sCi;

  const int row0 = blockIdx.y * BM;
  const int col0 = blockIdx.x * BN;
  const int tid  = threadIdx.x;
  const int warp = tid >> 5;
  const int lane = tid & 31;
  const int wm = warp >> 1;       // 0..3  (32-row slab)
  const int wn = warp & 1;        // 0..1  (64-col slab)
  const int lh = lane & 15;
  const int hi = lane >> 4;       // half-wave select

  const bool aOK = ((lda & 7) == 0) && ((((unsigned long)Ab) & 15) == 0);
  const bool bOK = ((ldb & 7) == 0) && ((((unsigned long)Bb) & 15) == 0);
  const int ktiles = (K + BKT - 1) / BKT;

  // ---- stage A tile [BM x BKT] into sA[buf] ----
  auto stageA = [&](int buf, int kt) {
    _Float16* dst = &sA[buf][0];
    const int kb = kt * BKT;
    if (aOK && (kb + BKT <= K)) {
      for (int i = tid; i < (BM * BKT) / 8; i += 256) {
        const int r = i >> 2;            // 4 vectors per 32-wide row
        const int c = (i & 3) * 8;
        const int gr = row0 + r;
        _Float16* l = dst + r * LDAS + c;
        if (gr < M) {
          const _Float16* p = Ab + (long)gr * lda + kb + c;
          copy16_g2l(p, l);
          __builtin_prefetch(p + 2 * BKT, 0, 1);  // warm k+2 slab
        } else {
          uint4 zz = {0u, 0u, 0u, 0u};
          *(uint4*)l = zz;
        }
      }
    } else {
      for (int i = tid; i < BM * BKT; i += 256) {
        const int r = i >> 5, c = i & 31;
        const int gr = row0 + r, gc = kb + c;
        _Float16 v = (_Float16)0.f;
        if (gr < M && gc < K) v = Ab[(long)gr * lda + gc];
        dst[r * LDAS + c] = v;
      }
    }
  };

  // ---- stage B tile as [BN(n) x BKT(k)] into sB[buf] ----
  auto stageB = [&](int buf, int kt) {
    _Float16* dst = &sB[buf][0];
    const int kb = kt * BKT;
    if (OPB == 1) {
      if (bOK && (kb + BKT <= K)) {
        for (int i = tid; i < (BN * BKT) / 8; i += 256) {
          const int r = i >> 2;
          const int c = (i & 3) * 8;
          const int gn = col0 + r;
          _Float16* l = dst + r * LDBS + c;
          if (gn < N) {
            const _Float16* p = Bb + (long)gn * ldb + kb + c;
            copy16_g2l(p, l);
            __builtin_prefetch(p + 2 * BKT, 0, 1);
          } else {
            uint4 zz = {0u, 0u, 0u, 0u};
            *(uint4*)l = zz;
          }
        }
      } else {
        for (int i = tid; i < BN * BKT; i += 256) {
          const int r = i >> 5, c = i & 31;
          const int gn = col0 + r, gc = kb + c;
          _Float16 v = (_Float16)0.f;
          if (gn < N && gc < K) v = Bb[(long)gn * ldb + gc];
          dst[r * LDBS + c] = v;
        }
      }
    } else {
      // B[K,N] row-major: vector load along N, scatter-transpose into LDS.
      for (int i = tid; i < (BN / 8) * BKT; i += 256) {
        const int k  = i >> 4;           // BN/8 == 16 vectors per k-row
        const int nv = (i & 15) * 8;
        const int gk = kb + k;
        _Float16 h[8];
        if (bOK && (gk < K) && (col0 + nv + 8 <= N)) {
          const _Float16* p = Bb + (long)gk * ldb + col0 + nv;
          *(uint4*)h = *(const uint4*)p;
          __builtin_prefetch(p + (long)2 * BKT * ldb, 0, 1);
        } else {
#pragma unroll
          for (int j = 0; j < 8; ++j) {
            const int gn = col0 + nv + j;
            h[j] = (gk < K && gn < N) ? Bb[(long)gk * ldb + gn] : (_Float16)0.f;
          }
        }
#pragma unroll
        for (int j = 0; j < 8; ++j) dst[(nv + j) * LDBS + k] = h[j];
      }
    }
  };

  v8f acc[2][4];
#pragma unroll
  for (int mi = 0; mi < 2; ++mi)
#pragma unroll
    for (int ni = 0; ni < 4; ++ni)
#pragma unroll
      for (int e = 0; e < 8; ++e) acc[mi][ni][e] = 0.0f;

  stageA(0, 0);
  stageB(0, 0);
  async_wait_all();
  __syncthreads();

  for (int kt = 0; kt < ktiles; ++kt) {
    const int cur = kt & 1;
    if (kt + 1 < ktiles) {        // stage next tile while computing this one
      stageA(cur ^ 1, kt + 1);
      stageB(cur ^ 1, kt + 1);
    }

    const _Float16* cA = &sA[cur][0];
    const _Float16* cB = &sB[cur][0];
    FragH af[2], bf[4];
#pragma unroll
    for (int mi = 0; mi < 2; ++mi) {
      const int row = wm * 32 + mi * 16 + lh;
      const int base = row * LDAS + hi * 8;        // A: K = hi*8 + [0..7], +16
      af[mi].q[0] = *(const uint4*)(&cA[base]);
      af[mi].q[1] = *(const uint4*)(&cA[base + 16]);
    }
#pragma unroll
    for (int ni = 0; ni < 4; ++ni) {
      const int col = wn * 64 + ni * 16 + lh;
      const int base = col * LDBS + hi * 8;        // B: K = hi*8 + [0..7], +16
      bf[ni].q[0] = *(const uint4*)(&cB[base]);
      bf[ni].q[1] = *(const uint4*)(&cB[base + 16]);
    }
#pragma unroll
    for (int mi = 0; mi < 2; ++mi)
#pragma unroll
      for (int ni = 0; ni < 4; ++ni)
        acc[mi][ni] = __builtin_amdgcn_wmma_f32_16x16x32_f16(
            false, af[mi].v, false, bf[ni].v, (short)0, acc[mi][ni], false, false);

    async_wait_all();             // next tile's async LDS writes landed
    __syncthreads();
  }

  // ---- epilogue ----
#pragma unroll
  for (int mi = 0; mi < 2; ++mi) {
    const int rbase = row0 + wm * 32 + mi * 16 + hi * 8;
#pragma unroll
    for (int ni = 0; ni < 4; ++ni) {
      const int col = col0 + wn * 64 + ni * 16 + lh;
      if (col < N) {
        const float bv = bias ? bias[col] : 0.0f;
#pragma unroll
        for (int e = 0; e < 8; ++e) {
          const int r = rbase + e;
          if (r < M) {
            float v = acc[mi][ni][e] + bv;
            if (RELU) v = fmaxf(v, 0.0f);
            const long idx = (long)r * ldc + col;
            if (ACCUM) Cb[idx] += (TC)v;
            else       Cb[idx]  = (TC)v;
          }
        }
      }
    }
  }
}

// ------------------------- small kernels -----------------------------------
__global__ void k_cast_f16(const float* __restrict__ s, _Float16* __restrict__ d, long n) {
  long i = (long)blockIdx.x * blockDim.x + threadIdx.x;
  if (i < n) d[i] = (_Float16)s[i];
}

// EnergyLayerNorm: scalar gamma, per-dim bias, RMS of centered x. f16 out.
__global__ __launch_bounds__(256) void k_enorm(
    const float* __restrict__ t, _Float16* __restrict__ g,
    const float* __restrict__ gamma, const float* __restrict__ bias, int D)
{
  __shared__ float red[256];
  const long row = blockIdx.x;
  const float* x = t + row * D;
  float s = 0.f;
  for (int d = threadIdx.x; d < D; d += 256) s += x[d];
  red[threadIdx.x] = s; __syncthreads();
  for (int o = 128; o > 0; o >>= 1) {
    if (threadIdx.x < o) red[threadIdx.x] += red[threadIdx.x + o];
    __syncthreads();
  }
  const float mean = red[0] / D;
  __syncthreads();
  float s2 = 0.f;
  for (int d = threadIdx.x; d < D; d += 256) { float v = x[d] - mean; s2 += v * v; }
  red[threadIdx.x] = s2; __syncthreads();
  for (int o = 128; o > 0; o >>= 1) {
    if (threadIdx.x < o) red[threadIdx.x] += red[threadIdx.x + o];
    __syncthreads();
  }
  const float r = rsqrtf(red[0] / D + 1e-5f);
  const float gm = gamma[0];
  for (int d = threadIdx.x; d < D; d += 256)
    g[row * D + d] = (_Float16)(gm * (x[d] - mean) * r + bias[d]);
}

// Standard LayerNorm (per-dim gain+bias), f16 out for decode GEMM.
__global__ __launch_bounds__(256) void k_layernorm(
    const float* __restrict__ t, _Float16* __restrict__ h,
    const float* __restrict__ gw, const float* __restrict__ bw, int D)
{
  __shared__ float red[256];
  const long row = blockIdx.x;
  const float* x = t + row * D;
  float s = 0.f;
  for (int d = threadIdx.x; d < D; d += 256) s += x[d];
  red[threadIdx.x] = s; __syncthreads();
  for (int o = 128; o > 0; o >>= 1) {
    if (threadIdx.x < o) red[threadIdx.x] += red[threadIdx.x + o];
    __syncthreads();
  }
  const float mean = red[0] / D;
  __syncthreads();
  float s2 = 0.f;
  for (int d = threadIdx.x; d < D; d += 256) { float v = x[d] - mean; s2 += v * v; }
  red[threadIdx.x] = s2; __syncthreads();
  for (int o = 128; o > 0; o >>= 1) {
    if (threadIdx.x < o) red[threadIdx.x] += red[threadIdx.x + o];
    __syncthreads();
  }
  const float r = rsqrtf(red[0] / D + 1e-5f);
  for (int d = threadIdx.x; d < D; d += 256)
    h[row * D + d] = (_Float16)((x[d] - mean) * r * gw[d] + bw[d]);
}

// row softmax of beta*S; writes P and P^T in f16.  One block per (bh, q) row.
__global__ __launch_bounds__(256) void k_softmax(
    const float* __restrict__ S, _Float16* __restrict__ P,
    _Float16* __restrict__ PT, int n, float beta)
{
  __shared__ float red[256];
  const long rowId = blockIdx.x;
  const long bh = rowId / n;
  const int  q  = (int)(rowId % n);
  const float* srow = S + bh * (long)n * n + (long)q * n;
  const int i = threadIdx.x;
  const float x = (i < n) ? beta * srow[i] : -3.4e38f;
  red[i] = x; __syncthreads();
  for (int o = 128; o > 0; o >>= 1) {
    if (i < o) red[i] = fmaxf(red[i], red[i + o]);
    __syncthreads();
  }
  const float mx = red[0]; __syncthreads();
  const float e = (i < n) ? __expf(x - mx) : 0.f;
  red[i] = e; __syncthreads();
  for (int o = 128; o > 0; o >>= 1) {
    if (i < o) red[i] += red[i + o];
    __syncthreads();
  }
  const float inv = 1.f / red[0];
  if (i < n) {
    const _Float16 p = (_Float16)(e * inv);
    P [bh * (long)n * n + (long)q * n + i] = p;
    PT[bh * (long)n * n + (long)i * n + q] = p;
  }
}

// t[b,0,:] = cls+pos[0];  t[b,1+n,:] = enc[b,n,:] + pos[1+n]  (b_enc in GEMM)
__global__ void k_assemble(const float* __restrict__ enc, const float* __restrict__ cls,
                           const float* __restrict__ pos, float* __restrict__ t)
{
  const long i = (long)blockIdx.x * blockDim.x + threadIdx.x;
  const long total = (long)CB * CNP * CTK;
  if (i >= total) return;
  const int d = (int)(i % CTK);
  const long rn = i / CTK;
  const int nn = (int)(rn % CNP);
  const long b = rn / CNP;
  float v;
  if (nn == 0) v = cls[d] + pos[d];
  else         v = enc[(b * CN + (nn - 1)) * CTK + d] + pos[(long)nn * CTK + d];
  t[i] = v;
}

// t += softplus(alpha_raw[blk]) * grad
__global__ void k_update(float* __restrict__ t, const float* __restrict__ grad,
                         const float* __restrict__ alpha_raw, int blk, long n)
{
  const long i = (long)blockIdx.x * blockDim.x + threadIdx.x;
  if (i < n) {
    const float a = log1pf(__expf(alpha_raw[blk]));
    t[i] += a * grad[i];
  }
}

// drop CLS row
__global__ void k_strip(const float* __restrict__ dec, float* __restrict__ out)
{
  const long i = (long)blockIdx.x * blockDim.x + threadIdx.x;
  const long total = (long)CB * CN * CD;
  if (i >= total) return;
  const int d = (int)(i % CD);
  const long rn = i / CD;
  const int nn = (int)(rn % CN);
  const long b = rn / CN;
  out[i] = dec[((b * CNP) + 1 + nn) * (long)CD + d];
}

// ------------------------- host launcher -----------------------------------
static inline dim3 ggrid(long M, long N, int Z) {
  return dim3((unsigned)((N + BN - 1) / BN), (unsigned)((M + BM - 1) / BM), (unsigned)Z);
}
static inline unsigned eblocks(long n) { return (unsigned)((n + 255) / 256); }

extern "C" void kernel_launch(void* const* d_in, const int* in_sizes, int n_in,
                              void* d_out, int out_size, void* d_ws, size_t ws_size,
                              hipStream_t stream) {
  const float* x     = (const float*)d_in[0];
  const float* W_enc = (const float*)d_in[1];
  const float* b_enc = (const float*)d_in[2];
  const float* cls   = (const float*)d_in[3];
  const float* pos   = (const float*)d_in[4];
  const float* gamma = (const float*)d_in[5];
  const float* biasE = (const float*)d_in[6];
  const float* Wq    = (const float*)d_in[7];
  const float* Wk    = (const float*)d_in[8];
  const float* Whn   = (const float*)d_in[9];
  const float* alpha = (const float*)d_in[10];
  const float* ln_g  = (const float*)d_in[11];
  const float* ln_b  = (const float*)d_in[12];
  const float* W_dec = (const float*)d_in[13];
  const float* b_dec = (const float*)d_in[14];
  float* out = (float*)d_out;

  // ---- workspace carve-out ----
  char* wsb = (char*)d_ws;
  size_t off = 0;
  auto take = [&](size_t bytes) -> char* {
    off = (off + 255) & ~(size_t)255;
    char* p = wsb + off;
    off += bytes;
    return p;
  };
  const long NB = (long)CB * CH;            // 768 batched heads
  const long SN = (long)CNP * CNP;          // 38809
  _Float16* xh    = (_Float16*)take(MENC * CTK * 2);
  _Float16* Wench = (_Float16*)take((long)CD * CTK * 2);
  _Float16* Wqh   = (_Float16*)take(2L * CTK * CTK * 2);
  _Float16* Wkh   = (_Float16*)take(2L * CTK * CTK * 2);
  _Float16* Whnh  = (_Float16*)take(2L * CTK * CHN * 2);
  _Float16* Wdech = (_Float16*)take((long)CTK * CD * 2);
  float*    t     = (float*)   take(MFULL * CTK * 4);
  _Float16* gh    = (_Float16*)take(MFULL * CTK * 2);
  _Float16* Qh    = (_Float16*)take(MFULL * CTK * 2);
  _Float16* Kh    = (_Float16*)take(MFULL * CTK * 2);
  _Float16* Ph    = (_Float16*)take(NB * SN * 2);
  _Float16* PTh   = (_Float16*)take(NB * SN * 2);
  _Float16* dqh   = (_Float16*)take(MFULL * CTK * 2);
  _Float16* dkh   = (_Float16*)take(MFULL * CTK * 2);
  char*     scr   = take(NB * SN * 4);      // aliases: S (f32) then H1 (f16)
  float*    Sbuf  = (float*)scr;
  _Float16* H1h   = (_Float16*)scr;
  float*    grad  = (float*)take(MFULL * CTK * 4);  // aliases enc_tmp/dec_tmp
  float*    enc_tmp = grad;
  float*    dec_tmp = grad;

  // ---- f32 -> f16 casts of inputs/weights ----
  k_cast_f16<<<eblocks(MENC * CTK), 256, 0, stream>>>(x, xh, MENC * CTK);
  k_cast_f16<<<eblocks((long)CD * CTK), 256, 0, stream>>>(W_enc, Wench, (long)CD * CTK);
  k_cast_f16<<<eblocks(2L * CTK * CTK), 256, 0, stream>>>(Wq, Wqh, 2L * CTK * CTK);
  k_cast_f16<<<eblocks(2L * CTK * CTK), 256, 0, stream>>>(Wk, Wkh, 2L * CTK * CTK);
  k_cast_f16<<<eblocks(2L * CTK * CHN), 256, 0, stream>>>(Whn, Whnh, 2L * CTK * CHN);
  k_cast_f16<<<eblocks((long)CTK * CD), 256, 0, stream>>>(W_dec, Wdech, (long)CTK * CD);

  // ---- encode: enc_tmp = x @ W_enc + b_enc ; then assemble t ----
  gemm_wmma<0, float, false, false><<<ggrid(MENC, CTK, 1), 256, 0, stream>>>(
      xh, Wench, enc_tmp, b_enc, (int)MENC, CTK, CD, CD, CTK, CTK,
      0, 0, 0, 0, 0, 0, 1);
  k_assemble<<<eblocks(MFULL * CTK), 256, 0, stream>>>(enc_tmp, cls, pos, t);

  // ---- ET blocks ----
  for (int blk = 0; blk < 2; ++blk) {
    const _Float16* wq  = Wqh  + (long)blk * CTK * CTK;
    const _Float16* wk  = Wkh  + (long)blk * CTK * CTK;
    const _Float16* whn = Whnh + (long)blk * CTK * CHN;
    for (int step = 0; step < 4; ++step) {
      // g = enorm(t)
      k_enorm<<<(unsigned)MFULL, 256, 0, stream>>>(t, gh, gamma + blk, biasE + blk * CTK, CTK);
      // Q = g Wq^T ; K = g Wk^T   (W stored [H*QK, TKN] = [N,K])
      gemm_wmma<1, _Float16, false, false><<<ggrid(MFULL, CTK, 1), 256, 0, stream>>>(
          gh, wq, Qh, nullptr, (int)MFULL, CTK, CTK, CTK, CTK, CTK, 0, 0, 0, 0, 0, 0, 1);
      gemm_wmma<1, _Float16, false, false><<<ggrid(MFULL, CTK, 1), 256, 0, stream>>>(
          gh, wk, Kh, nullptr, (int)MFULL, CTK, CTK, CTK, CTK, CTK, 0, 0, 0, 0, 0, 0, 1);
      // S[bh] = Q_bh K_bh^T   (batched over b,h; z = b*H + h)
      gemm_wmma<1, float, false, false><<<ggrid(CNP, CNP, (int)NB), 256, 0, stream>>>(
          Qh, Kh, Sbuf, nullptr, CNP, CNP, CQK, CTK, CTK, CNP,
          (long)CNP * CTK, CQK, (long)CNP * CTK, CQK, (long)CH * SN, SN, CH);
      // P = softmax(beta S) ; also P^T
      k_softmax<<<(unsigned)(NB * CNP), 256, 0, stream>>>(Sbuf, Ph, PTh, CNP, CBETA);
      // dq_bh = P K_bh ; dk_bh = P^T Q_bh   (write into packed [M, H*64])
      gemm_wmma<0, _Float16, false, false><<<ggrid(CNP, CQK, (int)NB), 256, 0, stream>>>(
          Ph, Kh, dqh, nullptr, CNP, CQK, CNP, CNP, CTK, CTK,
          (long)CH * SN, SN, (long)CNP * CTK, CQK, (long)CNP * CTK, CQK, CH);
      gemm_wmma<0, _Float16, false, false><<<ggrid(CNP, CQK, (int)NB), 256, 0, stream>>>(
          PTh, Qh, dkh, nullptr, CNP, CQK, CNP, CNP, CTK, CTK,
          (long)CH * SN, SN, (long)CNP * CTK, CQK, (long)CNP * CTK, CQK, CH);
      // grad = dq Wq ; grad += dk Wk   (B = W[c,d] row-major -> NN)
      gemm_wmma<0, float, false, false><<<ggrid(MFULL, CTK, 1), 256, 0, stream>>>(
          dqh, wq, grad, nullptr, (int)MFULL, CTK, CTK, CTK, CTK, CTK, 0, 0, 0, 0, 0, 0, 1);
      gemm_wmma<0, float, false, true><<<ggrid(MFULL, CTK, 1), 256, 0, stream>>>(
          dkh, wk, grad, nullptr, (int)MFULL, CTK, CTK, CTK, CTK, CTK, 0, 0, 0, 0, 0, 0, 1);
      // H1 = relu(g Whn)  [M, 3072] ; grad += H1 Whn^T
      gemm_wmma<0, _Float16, true, false><<<ggrid(MFULL, CHN, 1), 256, 0, stream>>>(
          gh, whn, H1h, nullptr, (int)MFULL, CHN, CTK, CTK, CHN, CHN, 0, 0, 0, 0, 0, 0, 1);
      gemm_wmma<1, float, false, true><<<ggrid(MFULL, CTK, 1), 256, 0, stream>>>(
          H1h, whn, grad, nullptr, (int)MFULL, CTK, CHN, CHN, CHN, CTK, 0, 0, 0, 0, 0, 0, 1);
      // t += softplus(alpha) * grad
      k_update<<<eblocks(MFULL * CTK), 256, 0, stream>>>(t, grad, alpha, blk, MFULL * CTK);
    }
  }

  // ---- decode: LN -> GEMM -> drop CLS ----
  k_layernorm<<<(unsigned)MFULL, 256, 0, stream>>>(t, gh, ln_g, ln_b, CTK);
  gemm_wmma<0, float, false, false><<<ggrid(MFULL, CD, 1), 256, 0, stream>>>(
      gh, Wdech, dec_tmp, b_dec, (int)MFULL, CD, CTK, CTK, CD, CD, 0, 0, 0, 0, 0, 0, 1);
  k_strip<<<eblocks((long)CB * CN * CD), 256, 0, stream>>>(dec_tmp, out);
}
